// SparseTransformerBlock_30751965839834
// MI455X (gfx1250) — matrix operand order repaired and checked by reading, not compiled, not run on hardware
//
#include <hip/hip_runtime.h>
#include <hip/hip_bf16.h>
#include <stdint.h>

typedef unsigned short u16;
typedef __attribute__((ext_vector_type(16))) __bf16 bf16x16;
typedef __attribute__((ext_vector_type(8)))  float  floatx8;
typedef int v4i __attribute__((vector_size(16)));   // b128 payload type for async-LDS builtins

#define N_TOK   16384
#define C_DIM   1024
#define SHIFT_S 256

#define AS_GLOBAL __attribute__((address_space(1)))
#define AS_SHARED __attribute__((address_space(3)))

#if defined(__AMDGCN__) && __has_builtin(__builtin_amdgcn_global_load_async_to_lds_b128) && __has_builtin(__builtin_amdgcn_s_wait_asynccnt)
#define USE_ASYNC_LDS 1
#else
#define USE_ASYNC_LDS 0
#endif

__device__ __forceinline__ u16 f2bf(float f) {
  union { float f; unsigned u; } c; c.f = f;
  unsigned u = c.u;
  u += 0x7FFFu + ((u >> 16) & 1u);       // round-to-nearest-even
  return (u16)(u >> 16);
}

__device__ __forceinline__ float gelu_tanh(float x) {
  float x3 = x * x * x;
  return 0.5f * x * (1.0f + tanhf(0.7978845608028654f * (x + 0.044715f * x3)));
}

// async (or fallback) copy of NCHUNK x 16 bytes from global bf16 to LDS bf16.
// Async-LDS builtin requires an immediate offset -> literal offsets per chunk.
template<int NCHUNK>
__device__ __forceinline__ void cp_g2s_16B(const u16* src, u16* dst) {
#if USE_ASYNC_LDS
  u16* sm = const_cast<u16*>(src);
  __builtin_amdgcn_global_load_async_to_lds_b128(
      (AS_GLOBAL v4i*)sm, (AS_SHARED v4i*)dst, 0, 0);
  if constexpr (NCHUNK > 1)
    __builtin_amdgcn_global_load_async_to_lds_b128(
        (AS_GLOBAL v4i*)sm, (AS_SHARED v4i*)dst, 16, 0);
  if constexpr (NCHUNK > 2)
    __builtin_amdgcn_global_load_async_to_lds_b128(
        (AS_GLOBAL v4i*)sm, (AS_SHARED v4i*)dst, 32, 0);
  if constexpr (NCHUNK > 3)
    __builtin_amdgcn_global_load_async_to_lds_b128(
        (AS_GLOBAL v4i*)sm, (AS_SHARED v4i*)dst, 48, 0);
#else
  #pragma unroll
  for (int c = 0; c < NCHUNK; ++c)
    ((uint4*)dst)[c] = ((const uint4*)src)[c];
#endif
}

__device__ __forceinline__ void wait_async_then_barrier() {
#if USE_ASYNC_LDS
  __builtin_amdgcn_s_wait_asynccnt(0);
#endif
  __syncthreads();
}

union F16x16 { uint4 q[2]; bf16x16 v; };

// ---------------------------------------------------------------------------
// Weight fp32 [K][N] -> bf16 transposed [N][K] (done once per launch; weights
// are reused by 128 M-tiles, so hot-loop conversion VALU is eliminated).
// ---------------------------------------------------------------------------
__global__ __launch_bounds__(256)
void wt_convert_kernel(const float* __restrict__ W, u16* __restrict__ Wt, int K, int N)
{
  __shared__ float tile[32][33];
  const int tx = threadIdx.x & 31;
  const int ty = threadIdx.x >> 5;            // 0..7
  const int n0 = blockIdx.x * 32;
  const int k0 = blockIdx.y * 32;
  #pragma unroll
  for (int j = 0; j < 4; ++j)
    tile[ty + j * 8][tx] = W[(size_t)(k0 + ty + j * 8) * N + n0 + tx];
  __syncthreads();
  #pragma unroll
  for (int j = 0; j < 4; ++j)
    Wt[(size_t)(n0 + ty + j * 8) * K + k0 + tx] = f2bf(tile[tx][ty + j * 8]);
}

// ---------------------------------------------------------------------------
// LayerNorm (fp32 stats, no affine) -> bf16, with optional source-row roll.
// ---------------------------------------------------------------------------
__global__ __launch_bounds__(256)
void ln_bf16_kernel(const float* __restrict__ x, u16* __restrict__ outp, int shift)
{
  const int row = blockIdx.x;
  int srow = row + shift;
  if (srow >= N_TOK) srow -= N_TOK;
  const int t = threadIdx.x;
  const float4 v = ((const float4*)(x + (size_t)srow * C_DIM))[t];
  float s = v.x + v.y + v.z + v.w;
  float q = v.x*v.x + v.y*v.y + v.z*v.z + v.w*v.w;
  #pragma unroll
  for (int off = 1; off < 32; off <<= 1) {
    s += __shfl_xor(s, off, 32);
    q += __shfl_xor(q, off, 32);
  }
  __shared__ float ss[8], sq[8];
  const int wid = t >> 5, lane = t & 31;
  if (lane == 0) { ss[wid] = s; sq[wid] = q; }
  __syncthreads();
  float ts = 0.0f, tq = 0.0f;
  #pragma unroll
  for (int i = 0; i < 8; ++i) { ts += ss[i]; tq += sq[i]; }
  const float mean = ts * (1.0f / C_DIM);
  const float var  = tq * (1.0f / C_DIM) - mean * mean;
  const float rstd = rsqrtf(var + 1e-6f);
  alignas(8) u16 ob[4];
  ob[0] = f2bf((v.x - mean) * rstd);
  ob[1] = f2bf((v.y - mean) * rstd);
  ob[2] = f2bf((v.z - mean) * rstd);
  ob[3] = f2bf((v.w - mean) * rstd);
  ((uint2*)(outp + (size_t)row * C_DIM))[t] = *(uint2*)ob;
}

// ---------------------------------------------------------------------------
// Tiled bf16 WMMA GEMM:  out[M,N] = epilogue( A[M,K] @ B[K,N] + bias )
//   A: bf16 (optionally concat with two fp32 sources converted on the fly)
//   Bt: bf16 PRE-TRANSPOSED [N][K]  -> both stages are pure b128 async copies
// Tile 128x256x32, 256 threads / 8 waves, wave = 64x64 (4x4 WMMA frags):
// 16 wmma per 8 fragment (16 b128) LDS loads per wave per K-step.
// ---------------------------------------------------------------------------
template<bool CONCAT, bool FUSE_GELU, bool OUT_BF16, bool RESID>
__global__ __launch_bounds__(256)
void gemm_wmma_kernel(const u16*  __restrict__ A,
                      const float* __restrict__ Acat1,
                      const float* __restrict__ Acat2,
                      const u16*  __restrict__ Bt,
                      const float* __restrict__ bias,
                      const float* __restrict__ resid,
                      void* __restrict__ outp,
                      int M, int Ndim, int K, int lda, int out_shift)
{
  constexpr int LDS_STRIDE = 48;          // 32 + pad, keeps 16B alignment
  __shared__ u16 As[128 * LDS_STRIDE];    // [row][k]
  __shared__ u16 Bs[256 * LDS_STRIDE];    // [col][k]  (B fragment layout)

  const int t    = threadIdx.x;
  const int lane = t & 31;
  const int wid  = t >> 5;
  const int l16  = lane & 15;
  const bool hi  = lane >= 16;
  const int wm   = wid >> 2;              // 0..1  -> M offset wm*64
  const int wn   = wid & 3;               // 0..3  -> N offset wn*64
  const int bm0  = blockIdx.y * 128;
  const int bn0  = blockIdx.x * 256;

  const int ar = t >> 1;                  // A stage: row, 2 threads/row
  const int ak = (t & 1) << 4;            //          k sub-offset 0/16

  const int aoff = hi ? 8  : 0;           // A frag K offset per ISA layout
  const int boff = hi ? 8  : 0;           // B frag K offset per ISA layout

  floatx8 acc[4][4];
  #pragma unroll
  for (int mt = 0; mt < 4; ++mt)
    #pragma unroll
    for (int nt = 0; nt < 4; ++nt)
      #pragma unroll
      for (int v = 0; v < 8; ++v) acc[mt][nt][v] = 0.0f;

  for (int k0 = 0; k0 < K; k0 += 32) {
    // ---- stage A tile: 128 x 32 bf16 ----
    {
      u16* dst = &As[ar * LDS_STRIDE + ak];
      const int grow = bm0 + ar;
      if constexpr (CONCAT) {
        const int gk = k0 + ak;            // branch uniform across block
        if (gk < C_DIM) {
          cp_g2s_16B<2>(A + (size_t)grow * lda + gk, dst);
        } else {
          const float* s = (gk < C_DIM + 4096)
              ? (Acat1 + (size_t)grow * 4096 + (gk - C_DIM))
              : (Acat2 + (size_t)grow * 4096 + (gk - C_DIM - 4096));
          alignas(16) u16 tmp[16];
          #pragma unroll
          for (int j = 0; j < 4; ++j) {
            float4 f = ((const float4*)s)[j];
            tmp[4*j+0] = f2bf(f.x); tmp[4*j+1] = f2bf(f.y);
            tmp[4*j+2] = f2bf(f.z); tmp[4*j+3] = f2bf(f.w);
          }
          ((uint4*)dst)[0] = ((uint4*)tmp)[0];
          ((uint4*)dst)[1] = ((uint4*)tmp)[1];
        }
      } else {
        const u16* s = A + (size_t)grow * lda + k0 + ak;
#if !USE_ASYNC_LDS
        if (k0 + 32 < K) __builtin_prefetch(s + 32, 0, 1);
#endif
        cp_g2s_16B<2>(s, dst);
      }
    }
    // ---- stage B tile: 256 cols x 32 k bf16 from pre-transposed Bt ----
    {
      const u16* s = Bt + (size_t)(bn0 + t) * K + k0;
#if !USE_ASYNC_LDS
      if (k0 + 32 < K) __builtin_prefetch(s + 32, 0, 1);
#endif
      cp_g2s_16B<4>(s, &Bs[t * LDS_STRIDE]);
    }
    wait_async_then_barrier();

    F16x16 af[4];
    #pragma unroll
    for (int mt = 0; mt < 4; ++mt) {
      const u16* p = &As[(wm*64 + mt*16 + l16) * LDS_STRIDE + aoff];
      af[mt].q[0] = *(const uint4*)p;
      af[mt].q[1] = *(const uint4*)(p + 16);
    }
    #pragma unroll
    for (int nt = 0; nt < 4; ++nt) {
      F16x16 bfr;
      const u16* p = &Bs[(wn*64 + nt*16 + l16) * LDS_STRIDE + boff];
      bfr.q[0] = *(const uint4*)p;
      bfr.q[1] = *(const uint4*)(p + 16);
      #pragma unroll
      for (int mt = 0; mt < 4; ++mt)
        acc[mt][nt] = __builtin_amdgcn_wmma_f32_16x16x32_bf16(
            false, af[mt].v, false, bfr.v, (short)0, acc[mt][nt], false, false);
    }
    __syncthreads();
  }

  // ---- epilogue: bias (+GELU) (+residual w/ row roll) -> bf16 or fp32 ----
  #pragma unroll
  for (int mt = 0; mt < 4; ++mt) {
    #pragma unroll
    for (int nt = 0; nt < 4; ++nt) {
      const int col = bn0 + wn*64 + nt*16 + l16;
      const float bv = bias[col];
      #pragma unroll
      for (int v = 0; v < 8; ++v) {
        const int row = bm0 + wm*64 + mt*16 + (hi ? v + 8 : v);
        int orow = row + out_shift;
        if (orow >= M) orow -= M;
        float xv = acc[mt][nt][v] + bv;
        if constexpr (FUSE_GELU) xv = gelu_tanh(xv);
        if constexpr (RESID)     xv += resid[(size_t)orow * Ndim + col];
        if constexpr (OUT_BF16)
          ((u16*)outp)[(size_t)orow * Ndim + col] = f2bf(xv);
        else
          ((float*)outp)[(size_t)orow * Ndim + col] = xv;
      }
    }
  }
}

// ---------------------------------------------------------------------------
// Windowed flash attention (WS=512, H=16, D=64) on bf16 QKV, WMMA bf16.
// Block = (q-tile 128, head, window); 8 waves, each owns 16 query rows.
// ---------------------------------------------------------------------------
__global__ __launch_bounds__(256)
void attn_wmma_kernel(const u16* __restrict__ qkv, u16* __restrict__ outp)
{
  __shared__ u16 Qs[128 * 80];        // [q][d]
  __shared__ u16 Ks[128 * 80];        // [key][d]   (B layout for S=Q.K^T)
  __shared__ u16 Vs[64 * 144];        // [d][key]   (B layout for O=P.V)
  __shared__ u16 Ps[8 * 16 * 136];    // wave-private P tiles [row][key]

  const int t    = threadIdx.x;
  const int lane = t & 31;
  const int wid  = t >> 5;
  const int l16  = lane & 15;
  const bool hi  = lane >= 16;

  const int qt   = blockIdx.x;        // 0..3
  const int h    = blockIdx.y;        // 0..15
  const int w    = blockIdx.z;        // 0..31
  const int wrow = w * 512;

  const int r  = t >> 1;
  const int d0 = (t & 1) << 5;

  // stage Q tile (128 x 64), async copy
  cp_g2s_16B<4>(qkv + (size_t)(wrow + qt*128 + r) * 3072 + h*64 + d0,
                &Qs[r*80 + d0]);

  float m_run[8], l_run[8];
  floatx8 o[4];
  #pragma unroll
  for (int v = 0; v < 8; ++v) { m_run[v] = -3.0e38f; l_run[v] = 0.0f; }
  #pragma unroll
  for (int dt = 0; dt < 4; ++dt)
    #pragma unroll
    for (int v = 0; v < 8; ++v) o[dt][v] = 0.0f;

  const float scale = 0.125f;   // 1/sqrt(64)

  for (int kb = 0; kb < 4; ++kb) {
    __syncthreads();
    // stage K [key][d] (async) and V transposed [d][key] (manual)
    {
      const size_t grow = (size_t)(wrow + kb*128 + r) * 3072;
      cp_g2s_16B<4>(qkv + grow + 1024 + h*64 + d0, &Ks[r*80 + d0]);
      alignas(16) u16 tv[32];
      const uint4* sv = (const uint4*)(qkv + grow + 2048 + h*64 + d0);
      ((uint4*)tv)[0] = sv[0]; ((uint4*)tv)[1] = sv[1];
      ((uint4*)tv)[2] = sv[2]; ((uint4*)tv)[3] = sv[3];
      #pragma unroll
      for (int j = 0; j < 32; ++j)
        Vs[(d0 + j) * 144 + r] = tv[j];
    }
    wait_async_then_barrier();

    // S(16 x 128) = Q_rows . K^T  (two K=32 chunks over d=64)
    F16x16 qa[2];
    #pragma unroll
    for (int kc = 0; kc < 2; ++kc) {
      const u16* p = &Qs[(wid*16 + l16)*80 + kc*32 + (hi ? 8 : 0)];
      qa[kc].q[0] = *(const uint4*)p;
      qa[kc].q[1] = *(const uint4*)(p + 16);
    }
    floatx8 sacc[8];
    #pragma unroll
    for (int nt = 0; nt < 8; ++nt) {
      #pragma unroll
      for (int v = 0; v < 8; ++v) sacc[nt][v] = 0.0f;
      #pragma unroll
      for (int kc = 0; kc < 2; ++kc) {
        F16x16 kf;
        const u16* p = &Ks[(nt*16 + l16)*80 + kc*32 + (hi ? 8 : 0)];
        kf.q[0] = *(const uint4*)p;
        kf.q[1] = *(const uint4*)(p + 16);
        sacc[nt] = __builtin_amdgcn_wmma_f32_16x16x32_bf16(
            false, qa[kc].v, false, kf.v, (short)0, sacc[nt], false, false);
      }
    }

    // online softmax; rows live per 16-lane half exactly as in C layout
    u16* pbase = &Ps[wid * 16 * 136];
    #pragma unroll
    for (int v = 0; v < 8; ++v) {
      float mx = -3.0e38f;
      #pragma unroll
      for (int nt = 0; nt < 8; ++nt) mx = fmaxf(mx, sacc[nt][v]);
      #pragma unroll
      for (int off = 1; off < 16; off <<= 1)
        mx = fmaxf(mx, __shfl_xor(mx, off, 32));
      mx *= scale;
      const float mnew  = fmaxf(m_run[v], mx);
      const float alpha = __expf(m_run[v] - mnew);
      m_run[v] = mnew;
      const int prow = hi ? v + 8 : v;
      float rs = 0.0f;
      #pragma unroll
      for (int nt = 0; nt < 8; ++nt) {
        const float p = __expf(sacc[nt][v] * scale - mnew);
        rs += p;
        pbase[prow * 136 + nt*16 + l16] = f2bf(p);
      }
      #pragma unroll
      for (int off = 1; off < 16; off <<= 1)
        rs += __shfl_xor(rs, off, 32);
      l_run[v] = l_run[v] * alpha + rs;
      #pragma unroll
      for (int dt = 0; dt < 4; ++dt) o[dt][v] *= alpha;
    }

    // O += P . V   (4 K=32 chunks over 128 keys)
    #pragma unroll
    for (int kc = 0; kc < 4; ++kc) {
      F16x16 pa;
      const u16* p = &pbase[l16 * 136 + kc*32 + (hi ? 8 : 0)];
      pa.q[0] = *(const uint4*)p;
      pa.q[1] = *(const uint4*)(p + 16);
      #pragma unroll
      for (int dt = 0; dt < 4; ++dt) {
        F16x16 vf;
        const u16* pv = &Vs[(dt*16 + l16)*144 + kc*32 + (hi ? 8 : 0)];
        vf.q[0] = *(const uint4*)pv;
        vf.q[1] = *(const uint4*)(pv + 16);
        o[dt] = __builtin_amdgcn_wmma_f32_16x16x32_bf16(
            false, pa.v, false, vf.v, (short)0, o[dt], false, false);
      }
    }
  }

  // write O / l  (bf16, still in shifted token order)
  #pragma unroll
  for (int v = 0; v < 8; ++v) {
    const float inv = 1.0f / l_run[v];
    const int row = wrow + qt*128 + wid*16 + (hi ? v + 8 : v);
    #pragma unroll
    for (int dt = 0; dt < 4; ++dt) {
      const int col = h*64 + dt*16 + l16;
      outp[(size_t)row * 1024 + col] = f2bf(o[dt][v] * inv);
    }
  }
}

// ---------------------------------------------------------------------------
extern "C" void kernel_launch(void* const* d_in, const int* in_sizes, int n_in,
                              void* d_out, int out_size, void* d_ws, size_t ws_size,
                              hipStream_t stream)
{
  (void)in_sizes; (void)n_in; (void)out_size; (void)ws_size;
  const float* x         = (const float*)d_in[0];
  const float* mv_normal = (const float*)d_in[1];
  const float* mv_color  = (const float*)d_in[2];
  const float* qkv_w     = (const float*)d_in[3];
  const float* qkv_b     = (const float*)d_in[4];
  const float* proj_w    = (const float*)d_in[5];
  const float* proj_b    = (const float*)d_in[6];
  const float* mv_w1     = (const float*)d_in[7];
  const float* mv_b1     = (const float*)d_in[8];
  const float* mv_w2     = (const float*)d_in[9];
  const float* mv_b2     = (const float*)d_in[10];
  const float* mlp_w1    = (const float*)d_in[11];
  const float* mlp_b1    = (const float*)d_in[12];
  const float* mlp_w2    = (const float*)d_in[13];
  const float* mlp_b2    = (const float*)d_in[14];
  float* xout = (float*)d_out;

  char* ws = (char*)d_ws;
  u16* a_ln  = (u16*)ws;  ws += (size_t)N_TOK * 1024 * sizeof(u16);
  u16* qkv   = (u16*)ws;  ws += (size_t)N_TOK * 3072 * sizeof(u16);
  u16* attn  = (u16*)ws;  ws += (size_t)N_TOK * 1024 * sizeof(u16);
  u16* hbig  = (u16*)ws;  ws += (size_t)N_TOK * 4096 * sizeof(u16);
  u16* wt_qkv  = (u16*)ws; ws += (size_t)3072 * 1024 * sizeof(u16);
  u16* wt_proj = (u16*)ws; ws += (size_t)1024 * 1024 * sizeof(u16);
  u16* wt_mv1  = (u16*)ws; ws += (size_t)4096 * 9216 * sizeof(u16);
  u16* wt_mv2  = (u16*)ws; ws += (size_t)1024 * 4096 * sizeof(u16);
  u16* wt_mlp1 = (u16*)ws; ws += (size_t)4096 * 1024 * sizeof(u16);
  u16* wt_mlp2 = (u16*)ws;

  dim3 blk(256);

  // 0. one-shot weight convert + transpose (fp32 [K][N] -> bf16 [N][K])
  wt_convert_kernel<<<dim3(3072/32, 1024/32), blk, 0, stream>>>(qkv_w,  wt_qkv,  1024, 3072);
  wt_convert_kernel<<<dim3(1024/32, 1024/32), blk, 0, stream>>>(proj_w, wt_proj, 1024, 1024);
  wt_convert_kernel<<<dim3(4096/32, 9216/32), blk, 0, stream>>>(mv_w1,  wt_mv1,  9216, 4096);
  wt_convert_kernel<<<dim3(1024/32, 4096/32), blk, 0, stream>>>(mv_w2,  wt_mv2,  4096, 1024);
  wt_convert_kernel<<<dim3(4096/32, 1024/32), blk, 0, stream>>>(mlp_w1, wt_mlp1, 1024, 4096);
  wt_convert_kernel<<<dim3(1024/32, 4096/32), blk, 0, stream>>>(mlp_w2, wt_mlp2, 4096, 1024);

  // 1. a_ln = LN(roll(x, -SHIFT))  (bf16)
  ln_bf16_kernel<<<N_TOK, blk, 0, stream>>>(x, a_ln, SHIFT_S);

  // 2. qkv = a_ln @ qkv_w + qkv_b  (bf16)
  gemm_wmma_kernel<false,false,true,false><<<dim3(3072/256, N_TOK/128), blk, 0, stream>>>(
      a_ln, nullptr, nullptr, wt_qkv, qkv_b, nullptr, qkv,
      N_TOK, 3072, 1024, 1024, 0);

  // 3. windowed flash attention (shifted order)
  attn_wmma_kernel<<<dim3(4, 16, 32), blk, 0, stream>>>(qkv, attn);

  // 4. xout = x + roll(attn @ proj_w + proj_b, +SHIFT)   (fp32)
  gemm_wmma_kernel<false,false,false,true><<<dim3(1024/256, N_TOK/128), blk, 0, stream>>>(
      attn, nullptr, nullptr, wt_proj, proj_b, x, xout,
      N_TOK, 1024, 1024, 1024, SHIFT_S);

  // 5. a_ln = LN(xout)
  ln_bf16_kernel<<<N_TOK, blk, 0, stream>>>(xout, a_ln, 0);

  // 6. hbig = gelu(concat(a_ln, mv_normal, mv_color) @ mv_w1 + mv_b1)  (bf16)
  gemm_wmma_kernel<true,true,true,false><<<dim3(4096/256, N_TOK/128), blk, 0, stream>>>(
      a_ln, mv_normal, mv_color, wt_mv1, mv_b1, nullptr, hbig,
      N_TOK, 4096, 9216, 1024, 0);

  // 7. xout += hbig @ mv_w2 + mv_b2
  gemm_wmma_kernel<false,false,false,true><<<dim3(1024/256, N_TOK/128), blk, 0, stream>>>(
      hbig, nullptr, nullptr, wt_mv2, mv_b2, xout, xout,
      N_TOK, 1024, 4096, 4096, 0);

  // 8. a_ln = LN(xout)
  ln_bf16_kernel<<<N_TOK, blk, 0, stream>>>(xout, a_ln, 0);

  // 9. hbig = gelu(a_ln @ mlp_w1 + mlp_b1)  (bf16)
  gemm_wmma_kernel<false,true,true,false><<<dim3(4096/256, N_TOK/128), blk, 0, stream>>>(
      a_ln, nullptr, nullptr, wt_mlp1, mlp_b1, nullptr, hbig,
      N_TOK, 4096, 1024, 1024, 0);

  // 10. xout += hbig @ mlp_w2 + mlp_b2
  gemm_wmma_kernel<false,false,false,true><<<dim3(1024/256, N_TOK/128), blk, 0, stream>>>(
      hbig, nullptr, nullptr, wt_mlp2, mlp_b2, xout, xout,
      N_TOK, 1024, 4096, 4096, 0);
}
